// Ses2Seq_27719718928986
// MI455X (gfx1250) — matrix operand order repaired and checked by reading, not compiled, not run on hardware
//
#include <hip/hip_runtime.h>
#include <hip/hip_bf16.h>

// Problem constants (from reference)
#define LATENT 8838
#define NP     8960            // LATENT padded to a multiple of 128
#define NBATCH 8
#define NTIME  256
#define LDSTR  40              // LDS panel row stride in halves (32 data + 8 pad)

typedef __attribute__((ext_vector_type(16))) _Float16 v16h;
typedef __attribute__((ext_vector_type(8)))  _Float16 v8h;
typedef __attribute__((ext_vector_type(8)))  float    v8f;

// ---------------------------------------------------------------------------
// WMMA fragment loaders (wave32, 16-bit A 16x32 / B 32x16 layouts per ISA 7.12.2)
//  A (row-major src): lane L (l=L&15, hi=L>>4) holds row m0+l,
//    halves K = k0 + hi*8 + {0..7}  and  k0 + 16 + hi*8 + {0..7}
//  B (loaded from the TRANSPOSED copy, row-major): lane holds column n0+l,
//    halves K = k0 + hi*8 + {0..7}  and  k0 + 16 + hi*8 + {0..7}
// ---------------------------------------------------------------------------
__device__ __forceinline__ v16h frag_a(const _Float16* __restrict__ p,
                                       int row, int k0, int ld, int hi) {
    const _Float16* base = p + (size_t)row * ld + k0 + hi * 8;
    v8h lo = *(const v8h*)(base);
    v8h hh = *(const v8h*)(base + 16);
    return __builtin_shufflevector(lo, hh, 0,1,2,3,4,5,6,7,8,9,10,11,12,13,14,15);
}

__device__ __forceinline__ v16h frag_b(const _Float16* __restrict__ pt,
                                       int col, int k0, int ld, int hi) {
    const _Float16* base = pt + (size_t)col * ld + k0 + hi * 8;
    v8h lo = *(const v8h*)(base);
    v8h hh = *(const v8h*)(base + 16);
    return __builtin_shufflevector(lo, hh, 0,1,2,3,4,5,6,7,8,9,10,11,12,13,14,15);
}

// Same layouts, but fed from an LDS panel with padded row stride LDSTR
__device__ __forceinline__ v16h lds_frag_a(const _Float16* base, int row, int hi) {
    const _Float16* p = base + row * LDSTR + hi * 8;
    v8h lo = *(const v8h*)(p);
    v8h hh = *(const v8h*)(p + 16);
    return __builtin_shufflevector(lo, hh, 0,1,2,3,4,5,6,7,8,9,10,11,12,13,14,15);
}

__device__ __forceinline__ v16h lds_frag_b(const _Float16* base, int col, int hi) {
    const _Float16* p = base + col * LDSTR + hi * 8;
    v8h lo = *(const v8h*)(p);
    v8h hh = *(const v8h*)(p + 16);
    return __builtin_shufflevector(lo, hh, 0,1,2,3,4,5,6,7,8,9,10,11,12,13,14,15);
}

__device__ __forceinline__ v8f wmma_f16(v16h a, v16h b, v8f c) {
    return __builtin_amdgcn_wmma_f32_16x16x32_f16(false, a, false, b,
                                                  (short)0, c, false, false);
}

// Async 16-byte Global -> LDS copy (CDNA5, tracked with ASYNCcnt).
// dst = LDS byte offset (low 32 bits of a generic shared pointer), src = global.
__device__ __forceinline__ void async_copy_b128(unsigned lds_off, const void* src) {
    asm volatile("global_load_async_to_lds_b128 %0, %1, off"
                 :: "v"(lds_off), "v"(src) : "memory");
}
__device__ __forceinline__ void wait_async0() {
    asm volatile("s_wait_asynccnt 0x0" ::: "memory");
}

// ---------------------------------------------------------------------------
// Init: E = A - I padded to NP x NP (f16), plus transposed copy.
// ---------------------------------------------------------------------------
__global__ void init_E_kernel(const float* __restrict__ A,
                              _Float16* __restrict__ H,
                              _Float16* __restrict__ HT) {
    size_t idx = (size_t)blockIdx.x * 256 + threadIdx.x;
    if (idx >= (size_t)NP * NP) return;
    int i = (int)(idx / NP), j = (int)(idx % NP);
    float v = 0.f;
    if (i < LATENT && j < LATENT) {
        v = A[(size_t)i * LATENT + j];
        if (i == j) v -= 1.f;
    }
    _Float16 h = (_Float16)v;
    H[idx] = h;
    HT[(size_t)j * NP + i] = h;
}

// UT row j (j<256) will hold the vector A^j b (f16, length NP). Init row0 = b.
__global__ void init_UT_kernel(const float* __restrict__ Bvec,
                               _Float16* __restrict__ UT) {
    size_t idx = (size_t)blockIdx.x * 256 + threadIdx.x;
    if (idx >= (size_t)NTIME * NP) return;
    int j = (int)(idx / NP), i = (int)(idx % NP);
    float v = (j == 0 && i < LATENT) ? Bvec[i] : 0.f;
    UT[idx] = (_Float16)v;
}

// ---------------------------------------------------------------------------
// Squaring GEMM:  M' = M*M + 2M   (f16 E-matrices, f32 WMMA accumulation)
// Block 256 thr = 8 waves (2x4); wave tile 64x64 -> block tile 128x256.
// K-panels double-buffered in LDS via async Global->LDS DMA so the copy of
// panel k+1 overlaps the 16 WMMAs of panel k. Epilogue writes row-major and
// transposed f16 copies.
// ---------------------------------------------------------------------------
__global__ void __launch_bounds__(256)
wmma_square_kernel(const _Float16* __restrict__ H, const _Float16* __restrict__ HT,
                   _Float16* __restrict__ Hn, _Float16* __restrict__ HTn) {
    __shared__ _Float16 Al[2][128 * LDSTR];
    __shared__ _Float16 Bl[2][256 * LDSTR];

    const int tid  = threadIdx.x;
    const int wave = tid >> 5;
    const int lane = tid & 31;
    const int l    = lane & 15;
    const int hi   = lane >> 4;
    const int bm   = blockIdx.x * 128;
    const int bn   = blockIdx.y * 256;
    const int m0l  = (wave >> 2) * 64;   // wave tile origin inside block tile
    const int n0l  = (wave & 3) * 64;

    // Stage one 32-wide K panel (A: 128x32, B: 256x32) into LDS buffer `buf`.
    auto stage = [&](int buf, int k0) {
        const unsigned abase = (unsigned)(uintptr_t)&Al[buf][0];
        const unsigned bbase = (unsigned)(uintptr_t)&Bl[buf][0];
#pragma unroll
        for (int c = tid; c < 512; c += 256) {            // A: 512 x 16B chunks
            const int row = c >> 2, ko = (c & 3) * 8;
            async_copy_b128(abase + (unsigned)(row * LDSTR + ko) * 2,
                            H + (size_t)(bm + row) * NP + k0 + ko);
        }
#pragma unroll
        for (int c = tid; c < 1024; c += 256) {           // B: 1024 x 16B chunks
            const int col = c >> 2, ko = (c & 3) * 8;
            async_copy_b128(bbase + (unsigned)(col * LDSTR + ko) * 2,
                            HT + (size_t)(bn + col) * NP + k0 + ko);
        }
    };

    v8f acc[4][4] = {};
    stage(0, 0);
    wait_async0();
    __syncthreads();

    int buf = 0;
    for (int k0 = 0; k0 < NP; k0 += 32) {
        if (k0 + 32 < NP) stage(buf ^ 1, k0 + 32);        // overlap next panel
        v16h a[4], b[4];
#pragma unroll
        for (int i = 0; i < 4; ++i) a[i] = lds_frag_a(&Al[buf][0], m0l + 16 * i + l, hi);
#pragma unroll
        for (int j = 0; j < 4; ++j) b[j] = lds_frag_b(&Bl[buf][0], n0l + 16 * j + l, hi);
#pragma unroll
        for (int i = 0; i < 4; ++i)
#pragma unroll
            for (int j = 0; j < 4; ++j)
                acc[i][j] = wmma_f16(a[i], b[j], acc[i][j]);
        wait_async0();                                    // our next-panel DMAs done
        __syncthreads();                                  // everyone's reads + DMAs done
        buf ^= 1;
    }

    // D(row, col): row = bm+m0l+16i + r + 8*hi, col = bn+n0l+16j + l (ISA C/D layout)
#pragma unroll
    for (int i = 0; i < 4; ++i) {
#pragma unroll
        for (int j = 0; j < 4; ++j) {
            const int col = bn + n0l + 16 * j + l;
#pragma unroll
            for (int r = 0; r < 8; ++r) {
                const int row = bm + m0l + 16 * i + r + 8 * hi;
                float e = (float)H[(size_t)row * NP + col];
                _Float16 hv = (_Float16)(acc[i][j][r] + 2.f * e);
                Hn [(size_t)row * NP + col] = hv;
                HTn[(size_t)col * NP + row] = hv;
            }
        }
    }
}

// ---------------------------------------------------------------------------
// Krylov doubling: rows [m, 2m) of UT  =  rows [0,m) + M_k * rows [0,m).
// UT is the transposed Krylov block (row j = A^j b), so it IS the B operand.
// Block 128 thr = 4 waves; wave tile 64x16.
// ---------------------------------------------------------------------------
__global__ void __launch_bounds__(128)
wmma_krylov_kernel(const _Float16* __restrict__ H, _Float16* __restrict__ UT, int m) {
    const int tid  = threadIdx.x;
    const int wave = tid >> 5;
    const int lane = tid & 31;
    const int l    = lane & 15;
    const int hi   = lane >> 4;
    const int m0   = blockIdx.x * 256 + wave * 64;
    const int n0   = blockIdx.y * 16;

    v8f acc[4] = {};
    for (int k0 = 0; k0 < NP; k0 += 32) {
        v16h b = frag_b(UT, n0 + l, k0, NP, hi);
#pragma unroll
        for (int i = 0; i < 4; ++i) {
            v16h a = frag_a(H, m0 + 16 * i + l, k0, NP, hi);
            acc[i] = wmma_f16(a, b, acc[i]);
        }
    }
    const int j = n0 + l;
    if (j < m) {
#pragma unroll
        for (int i = 0; i < 4; ++i)
#pragma unroll
            for (int r = 0; r < 8; ++r) {
                const int row = m0 + 16 * i + r + 8 * hi;
                float u = (float)UT[(size_t)j * NP + row];
                UT[(size_t)(m + j) * NP + row] = (_Float16)(u + acc[i][r]);
            }
    }
}

// ---------------------------------------------------------------------------
// c = theta + M8 * theta   (batch-independent; one block of 256 per row)
// ---------------------------------------------------------------------------
__global__ void matvec_c_kernel(const _Float16* __restrict__ H8,
                                const float* __restrict__ theta,
                                float* __restrict__ c) {
    __shared__ float red[256];
    const int i = blockIdx.x;
    float s = 0.f;
    for (int k = threadIdx.x; k < LATENT; k += 256)
        s += (float)H8[(size_t)i * NP + k] * theta[k];
    red[threadIdx.x] = s;
    __syncthreads();
    for (int off = 128; off > 0; off >>= 1) {
        if (threadIdx.x < off) red[threadIdx.x] += red[threadIdx.x + off];
        __syncthreads();
    }
    if (threadIdx.x == 0)
        c[i] = ((i < LATENT) ? theta[i] : 0.f) + red[0];
}

// ---------------------------------------------------------------------------
// th[b, :] = c + sum_j UT[j, :] * x[b, 255-j]     (xs flat: x[b,t]=xs[b*256+t])
// ---------------------------------------------------------------------------
__global__ void __launch_bounds__(256)
krylov_apply_kernel(const _Float16* __restrict__ UT, const float* __restrict__ xs,
                    const float* __restrict__ c, float* __restrict__ th) {
    __shared__ float xl[NTIME * NBATCH];   // xl[t*8+b]
    for (int idx = threadIdx.x; idx < NTIME * NBATCH; idx += 256) {
        int b = idx / NTIME, t = idx % NTIME;
        xl[t * NBATCH + b] = xs[b * NTIME + t];
    }
    __syncthreads();
    const int i = blockIdx.x * 256 + threadIdx.x;
    float acc[NBATCH] = {};
    for (int j = 0; j < NTIME; ++j) {
        float u = (float)UT[(size_t)j * NP + i];
        const int t = 255 - j;
#pragma unroll
        for (int b = 0; b < NBATCH; ++b) acc[b] += u * xl[t * NBATCH + b];
    }
    float ci = c[i];
#pragma unroll
    for (int b = 0; b < NBATCH; ++b) th[(size_t)b * NP + i] = ci + acc[b];
}

// ---------------------------------------------------------------------------
// MLP epilogue: per batch, weights cached in LDS; one thread per time point.
// ---------------------------------------------------------------------------
__global__ void __launch_bounds__(256)
mlp_eval_kernel(const float* __restrict__ th, const float* __restrict__ ts,
                float* __restrict__ out) {
    __shared__ float w[LATENT];
    const int b = blockIdx.x, t = threadIdx.x;
    for (int idx = t; idx < LATENT; idx += 256) w[idx] = th[(size_t)b * NP + idx];
    __syncthreads();
    const float x = ts[b * NTIME + t];
    float h1[64], h2[64];
#pragma unroll
    for (int o = 0; o < 64; ++o) h1[o] = fmaxf(w[o] * x + w[64 + o], 0.f);
    for (int o = 0; o < 64; ++o) {
        float s = w[4224 + o];
        for (int d = 0; d < 64; ++d) s += w[128 + o * 64 + d] * h1[d];
        h2[o] = fmaxf(s, 0.f);
    }
    for (int o = 0; o < 64; ++o) {
        float s = w[8384 + o];
        for (int d = 0; d < 64; ++d) s += w[4288 + o * 64 + d] * h2[d];
        h1[o] = fmaxf(s, 0.f);
    }
    for (int o = 0; o < 6; ++o) {
        float s = w[8832 + o];
        for (int d = 0; d < 64; ++d) s += w[8448 + o * 64 + d] * h1[d];
        out[((size_t)b * NTIME + t) * 6 + o] = s;
    }
}

// ---------------------------------------------------------------------------
extern "C" void kernel_launch(void* const* d_in, const int* in_sizes, int n_in,
                              void* d_out, int out_size, void* d_ws, size_t ws_size,
                              hipStream_t stream) {
    const float* xs    = (const float*)d_in[0];   // (8,1,256,1)
    const float* ts    = (const float*)d_in[1];   // (8,256)
    const float* theta = (const float*)d_in[2];   // (8838,)
    const float* A     = (const float*)d_in[3];   // (8838,8838)
    const float* Bvec  = (const float*)d_in[4];   // (8838,1)
    float* out = (float*)d_out;

    // Workspace carve-up (~617 MB required)
    const size_t szH  = (size_t)NP * NP * sizeof(_Float16);   // 160.6 MB each
    const size_t szUT = (size_t)NTIME * NP * sizeof(_Float16);
    char* wsb = (char*)d_ws;
    _Float16* Hb[2]  = { (_Float16*)(wsb),            (_Float16*)(wsb + 2 * szH) };
    _Float16* HTb[2] = { (_Float16*)(wsb + szH),      (_Float16*)(wsb + 3 * szH) };
    _Float16* UT     = (_Float16*)(wsb + 4 * szH);
    float*    c      = (float*)   (wsb + 4 * szH + szUT);
    float*    th     = (float*)   (wsb + 4 * szH + szUT + (size_t)NP * sizeof(float));

    const size_t nE = (size_t)NP * NP;
    init_E_kernel <<<(unsigned)((nE + 255) / 256), 256, 0, stream>>>(A, Hb[0], HTb[0]);
    init_UT_kernel<<<(unsigned)(((size_t)NTIME * NP + 255) / 256), 256, 0, stream>>>(Bvec, UT);

    // 8 doubling stages: Krylov expand with M_k, then square M_k -> M_{k+1}
    for (int k = 0; k < 8; ++k) {
        const int m = 1 << k;
        wmma_krylov_kernel<<<dim3(NP / 256, (m + 15) / 16), 128, 0, stream>>>(
            Hb[k & 1], UT, m);
        wmma_square_kernel<<<dim3(NP / 128, NP / 256), 256, 0, stream>>>(
            Hb[k & 1], HTb[k & 1], Hb[(k + 1) & 1], HTb[(k + 1) & 1]);
    }
    // M8 = A^256 - I now lives in Hb[0]
    matvec_c_kernel    <<<NP, 256, 0, stream>>>(Hb[0], theta, c);
    krylov_apply_kernel<<<NP / 256, 256, 0, stream>>>(UT, xs, c, th);
    mlp_eval_kernel    <<<NBATCH, 256, 0, stream>>>(th, ts, out);

    (void)in_sizes; (void)n_in; (void)out_size; (void)ws_size;
}
